// NeighborhoodAggregation_47991964565560
// MI455X (gfx1250) — compile-verified, matrix operand order checked by reading, not run on hardware
//
#include <hip/hip_runtime.h>
#include <hip/hip_bf16.h>

// ---------------- problem constants (from reference) ----------------
#define B_ROWS 1024
#define D_DIM  1024
#define N_MEM  100000
#define C_CLS  1000
#define TOPK   5

// ---------------- tiling ----------------
#define MB     4                 // M blocks: 4 * 256 rows = 1024
#define NCH    64                // N chunks (partial top-k per chunk)
#define NTILES (N_MEM / 16)      // 6250 column tiles of 16
#define PER    ((NTILES + NCH - 1) / NCH)   // 98 tiles per chunk

typedef __attribute__((ext_vector_type(16))) _Float16 v16h;
typedef __attribute__((ext_vector_type(8)))  _Float16 v8h;
typedef __attribute__((ext_vector_type(4)))  _Float16 v4h;
typedef __attribute__((ext_vector_type(8)))  float    v8f;
typedef __attribute__((ext_vector_type(4)))  float    v4f;

// s_wait_dscnt has no clang builtin on this toolchain -> inline asm (SOPP op 70)
__device__ __forceinline__ void wait_dscnt0() {
  asm volatile("s_wait_dscnt 0" ::: "memory");
}

// ========== kernel 1: row-normalize features, emit f16 ==========
__global__ void na_normalize_f16(const float* __restrict__ feat,
                                 _Float16* __restrict__ x_h) {
  int row = blockIdx.x;
  int tid = threadIdx.x;
  const float* fr = feat + (size_t)row * D_DIM;
  v4f v = ((const v4f*)fr)[tid];                  // 256 threads * 4 = 1024
  float ss = v[0]*v[0] + v[1]*v[1] + v[2]*v[2] + v[3]*v[3];
  __shared__ float red[256];
  red[tid] = ss;
  __syncthreads();
  for (int s = 128; s > 0; s >>= 1) {
    if (tid < s) red[tid] += red[tid + s];
    __syncthreads();
  }
  float scale = 1.0f / fmaxf(sqrtf(red[0]), 1e-12f);
  v4h h;
  h[0] = (_Float16)(v[0] * scale);
  h[1] = (_Float16)(v[1] * scale);
  h[2] = (_Float16)(v[2] * scale);
  h[3] = (_Float16)(v[3] * scale);
  *(v4h*)(x_h + (size_t)row * D_DIM + tid * 4) = h;
}

// ========== kernel 1b: stream-convert feat_memory f32 -> f16 copy ==========
// Read-once stream: non-temporal loads keep L2 free for the f16 copy,
// which (205 MB vs 192 MB L2) stays mostly L2-resident for the GEMM passes.
__global__ void na_cvt_fm(const float* __restrict__ fm,
                          _Float16* __restrict__ fm_h, long n4) {
  long i = (long)blockIdx.x * blockDim.x + threadIdx.x;
  long stride = (long)gridDim.x * blockDim.x;
  for (; i < n4; i += stride) {
    v4f v = __builtin_nontemporal_load((const v4f*)fm + i);
    v4h h;
    h[0] = (_Float16)v[0];
    h[1] = (_Float16)v[1];
    h[2] = (_Float16)v[2];
    h[3] = (_Float16)v[3];
    *((v4h*)fm_h + i) = h;
  }
}

// ---- fragment loaders (CDNA5 16-bit 16x32 A/B layout) ----
// lane<16 : K = {k..k+7, k+16..k+23}; lane>=16 : K = {k+8..k+15, k+24..k+31}
__device__ __forceinline__ v16h frag_from_f16(const _Float16* rowp, int kk, int lane) {
  int k0 = kk + ((lane & 16) ? 8 : 0);
  v8h lo = *(const v8h*)(rowp + k0);
  v8h hi = *(const v8h*)(rowp + k0 + 16);
  v16h r;
#pragma unroll
  for (int i = 0; i < 8; ++i) { r[i] = lo[i]; r[i + 8] = hi[i]; }
  return r;
}

__device__ __forceinline__ v16h frag_from_f32(const float* rowp, int kk, int lane) {
  int k0 = kk + ((lane & 16) ? 8 : 0);
  v4f a0 = *(const v4f*)(rowp + k0);
  v4f a1 = *(const v4f*)(rowp + k0 + 4);
  v4f b0 = *(const v4f*)(rowp + k0 + 16);
  v4f b1 = *(const v4f*)(rowp + k0 + 20);
  v16h r;
#pragma unroll
  for (int i = 0; i < 4; ++i) {
    r[i]      = (_Float16)a0[i];
    r[i + 4]  = (_Float16)a1[i];
    r[i + 8]  = (_Float16)b0[i];
    r[i + 12] = (_Float16)b1[i];
  }
  return r;
}

__device__ __forceinline__ void topk_insert(float* tv, int* ti, float v, int id) {
  if (v <= tv[TOPK - 1]) return;
  int j = TOPK - 1;
#pragma unroll
  for (int s = 0; s < TOPK - 1; ++s) {
    if (j > 0 && v > tv[j - 1]) { tv[j] = tv[j - 1]; ti[j] = ti[j - 1]; --j; }
  }
  tv[j] = v; ti[j] = id;
}

// ========== kernel 2: fused WMMA GEMM + per-chunk top-5 ==========
// grid = (NCH, MB), block = 256 (8 waves). Each wave owns two 16-row M tiles
// (rows m0 + w*16 and m0 + 128 + w*16) and streams 16-column tiles of its chunk.
// BH = true : B comes from the pre-converted f16 copy (no cvt in hot loop).
template <bool BH>
__global__ __launch_bounds__(256)
void na_gemm_topk(const _Float16* __restrict__ x_h,
                  const void* __restrict__ fmv,
                  const int* __restrict__ idxp,
                  float* __restrict__ part_val,
                  int* __restrict__ part_idx) {
  const int chunk = blockIdx.x;
  const int m0    = blockIdx.y * 256;
  const int w     = threadIdx.x >> 5;
  const int lane  = threadIdx.x & 31;
  const int l15   = lane & 15;

  const int mrow0 = m0 + w * 16 + l15;        // tile 0 row for this lane
  const int mrow1 = m0 + 128 + w * 16 + l15;  // tile 1 row for this lane

  const _Float16* arow0 = x_h + (size_t)mrow0 * D_DIM;
  const _Float16* arow1 = x_h + (size_t)mrow1 * D_DIM;

  int selfI0 = idxp[mrow0];
  int selfI1 = idxp[mrow1];

  float tv[2][TOPK];
  int   ti[2][TOPK];
#pragma unroll
  for (int t = 0; t < 2; ++t)
#pragma unroll
    for (int j = 0; j < TOPK; ++j) { tv[t][j] = -3.0e38f; ti[t][j] = 0; }

  __shared__ float stile[8][2][16][17];   // per-wave C tile staging

  const int t0 = chunk * PER;
  const int t1 = (t0 + PER < NTILES) ? (t0 + PER) : NTILES;

  for (int tile = t0; tile < t1; ++tile) {
    const int n0 = tile * 16;

    v8f c0 = {};
    v8f c1 = {};
#pragma unroll 4
    for (int kk = 0; kk < D_DIM; kk += 32) {
      v16h b;
      if constexpr (BH) {
        b = frag_from_f16((const _Float16*)fmv + (size_t)(n0 + l15) * D_DIM, kk, lane);
      } else {
        b = frag_from_f32((const float*)fmv + (size_t)(n0 + l15) * D_DIM, kk, lane);
      }
      v16h a0 = frag_from_f16(arow0, kk, lane);
      v16h a1 = frag_from_f16(arow1, kk, lane);
      c0 = __builtin_amdgcn_wmma_f32_16x16x32_f16(false, a0, false, b,
                                                  (short)0, c0, false, false);
      c1 = __builtin_amdgcn_wmma_f32_16x16x32_f16(false, a1, false, b,
                                                  (short)0, c1, false, false);
    }

    // spill C tiles through LDS to get row-major per-lane access
    const int mh = (lane >> 4) ? 8 : 0;
#pragma unroll
    for (int r = 0; r < 8; ++r) {
      stile[w][0][r + mh][l15] = c0[r];
      stile[w][1][r + mh][l15] = c1[r];
    }
    // per-wave private LDS region: explicit wait for DS stores before reads
    wait_dscnt0();

    if (lane < 16) {
#pragma unroll
      for (int t = 0; t < 2; ++t) {
        const int self = t ? selfI1 : selfI0;
#pragma unroll
        for (int n = 0; n < 16; ++n) {
          float v = stile[w][t][lane][n];
          int col = n0 + n;
          if (col != self) topk_insert(tv[t], ti[t], v, col);
        }
      }
    }
    wait_dscnt0();
  }

  if (lane < 16) {
#pragma unroll
    for (int t = 0; t < 2; ++t) {
      int row = t ? mrow1 : mrow0;
      size_t base = ((size_t)row * NCH + chunk) * TOPK;
#pragma unroll
      for (int j = 0; j < TOPK; ++j) {
        part_val[base + j] = tv[t][j];
        part_idx[base + j] = ti[t][j];
      }
    }
  }
}

// ========== kernel 3: merge partials, gather pred_memory, mean, argmax ==========
__global__ __launch_bounds__(256)
void na_finalize(const float* __restrict__ part_val,
                 const int* __restrict__ part_idx,
                 const float* __restrict__ pred,
                 float* __restrict__ out_labels,
                 float* __restrict__ out_logits) {
  const int row = blockIdx.x;
  const int tid = threadIdx.x;
  __shared__ int sel[TOPK];
  __shared__ float rv[256];
  __shared__ int   ri[256];

  if (tid == 0) {
    float tv[TOPK]; int ti[TOPK];
#pragma unroll
    for (int j = 0; j < TOPK; ++j) { tv[j] = -3.0e38f; ti[j] = 0; }
    size_t base = (size_t)row * NCH * TOPK;
    for (int c = 0; c < NCH; ++c) {
      for (int j = 0; j < TOPK; ++j) {
        float v = part_val[base + c * TOPK + j];
        int  id = part_idx[base + c * TOPK + j];
        topk_insert(tv, ti, v, id);
      }
    }
#pragma unroll
    for (int j = 0; j < TOPK; ++j) sel[j] = ti[j];
  }
  __syncthreads();

  int s0 = sel[0], s1 = sel[1], s2 = sel[2], s3 = sel[3], s4 = sel[4];
  float bestv = -3.0e38f;
  int   besti = 0;
  for (int c = tid; c < C_CLS; c += 256) {
    float s = pred[(size_t)s0 * C_CLS + c] + pred[(size_t)s1 * C_CLS + c] +
              pred[(size_t)s2 * C_CLS + c] + pred[(size_t)s3 * C_CLS + c] +
              pred[(size_t)s4 * C_CLS + c];
    s *= 0.2f;
    out_logits[(size_t)row * C_CLS + c] = s;
    if (s > bestv) { bestv = s; besti = c; }
  }
  rv[tid] = bestv; ri[tid] = besti;
  __syncthreads();
  for (int s = 128; s > 0; s >>= 1) {
    if (tid < s) {
      float ov = rv[tid + s]; int oi = ri[tid + s];
      if (ov > rv[tid] || (ov == rv[tid] && oi < ri[tid])) { rv[tid] = ov; ri[tid] = oi; }
    }
    __syncthreads();
  }
  if (tid == 0) out_labels[row] = (float)ri[0];
}

// ========== launcher ==========
extern "C" void kernel_launch(void* const* d_in, const int* in_sizes, int n_in,
                              void* d_out, int out_size, void* d_ws, size_t ws_size,
                              hipStream_t stream) {
  const float* features    = (const float*)d_in[0];
  const int*   idxp        = (const int*)d_in[1];
  const float* feat_memory = (const float*)d_in[2];
  const float* pred_memory = (const float*)d_in[3];

  // workspace layout
  _Float16* x_h      = (_Float16*)d_ws;                              // 2 MB
  char*     p        = (char*)d_ws + (size_t)B_ROWS * D_DIM * 2;
  float*    part_val = (float*)p;                                    // 1.31 MB
  p += (size_t)B_ROWS * NCH * TOPK * sizeof(float);
  int*      part_idx = (int*)p;                                      // 1.31 MB
  p += (size_t)B_ROWS * NCH * TOPK * sizeof(int);
  size_t base_bytes = (size_t)(p - (char*)d_ws);                     // 4.72 MB
  size_t fm_bytes   = (size_t)N_MEM * D_DIM * sizeof(_Float16);      // 204.8 MB
  bool   use_f16_fm = (ws_size >= base_bytes + fm_bytes);

  float* out_labels = (float*)d_out;
  float* out_logits = out_labels + B_ROWS;

  na_normalize_f16<<<B_ROWS, 256, 0, stream>>>(features, x_h);

  dim3 grid(NCH, MB);
  if (use_f16_fm) {
    _Float16* fm_h = (_Float16*)((char*)d_ws + base_bytes);          // 16B aligned
    na_cvt_fm<<<2048, 256, 0, stream>>>(feat_memory, fm_h,
                                        (long)N_MEM * D_DIM / 4);
    na_gemm_topk<true><<<grid, 256, 0, stream>>>(x_h, fm_h, idxp,
                                                 part_val, part_idx);
  } else {
    na_gemm_topk<false><<<grid, 256, 0, stream>>>(x_h, feat_memory, idxp,
                                                  part_val, part_idx);
  }

  na_finalize<<<B_ROWS, 256, 0, stream>>>(part_val, part_idx, pred_memory,
                                          out_labels, out_logits);
}
